// HierarchicalMemoryAttention_54185307406570
// MI455X (gfx1250) — matrix operand order repaired and checked by reading, not compiled, not run on hardware
//
#include <hip/hip_runtime.h>
#include <hip/hip_bf16.h>

// Problem sizes (fixed by reference)
#define BB 4
#define QQ 256
#define EE 512
#define MM 1024
#define CC 32
#define DD 512
#define FF 512
#define HH 8
#define KTOP 8
#define HDIM 64

typedef __attribute__((ext_vector_type(16))) __bf16 v16bf;
typedef __attribute__((ext_vector_type(8)))  __bf16 v8bf;
typedef __attribute__((ext_vector_type(8)))  float  v8f;
typedef int v4i __attribute__((vector_size(16)));   // matches builtin's V4i pointee

#if __has_builtin(__builtin_amdgcn_global_load_async_to_lds_b128) && \
    __has_builtin(__builtin_amdgcn_s_wait_asynccnt)
#define HAS_ASYNC 1
#else
#define HAS_ASYNC 0
#endif

#define AS1 __attribute__((address_space(1)))
#define AS3 __attribute__((address_space(3)))

__device__ __forceinline__ __bf16 f2bf(float f) {
  unsigned u = __builtin_bit_cast(unsigned, f);
  unsigned r = (u + 0x7FFFu + ((u >> 16) & 1u)) >> 16;
  unsigned short s = (unsigned short)r;
  return __builtin_bit_cast(__bf16, s);
}
__device__ __forceinline__ float bf2f(__bf16 x) {
  unsigned short s = __builtin_bit_cast(unsigned short, x);
  unsigned u = (unsigned)s << 16;
  return __builtin_bit_cast(float, u);
}

// ---------------------------------------------------------------------------
// Elementwise fp32 -> bf16
// ---------------------------------------------------------------------------
__global__ void k_f32_to_bf16(const float* __restrict__ in, __bf16* __restrict__ out, size_t n) {
  size_t i = (size_t)blockIdx.x * blockDim.x + threadIdx.x;
  size_t stride = (size_t)gridDim.x * blockDim.x;
  for (; i < n; i += stride) out[i] = f2bf(in[i]);
}

// Transpose-convert weights: in [K,N] fp32 row-major -> out [N,K] bf16 row-major
__global__ void k_transpose_bf16(const float* __restrict__ in, __bf16* __restrict__ out,
                                 int Kd, int Nd) {
  int idx = blockIdx.x * blockDim.x + threadIdx.x;
  if (idx >= Kd * Nd) return;
  int n = idx / Kd, k = idx % Kd;
  out[idx] = f2bf(in[(size_t)k * Nd + n]);
}

// Sinusoid position encoding [C=32, D=512] -> bf16
__global__ void k_pos_encode(__bf16* __restrict__ out) {
  int idx = blockIdx.x * blockDim.x + threadIdx.x;
  if (idx >= CC * DD) return;
  int c = idx >> 9, d = idx & 511;
  float p = 31.0f - (float)c;            // pos_seq = [C-1 .. 0]
  int j = d & 255;                       // freqs = 2*j
  float inv = __powf(10000.0f, -(2.0f * (float)j) / 512.0f);
  float ang = p * inv;
  out[idx] = f2bf(d < 256 ? __sinf(ang) : __cosf(ang));
}

// ---------------------------------------------------------------------------
// NT bf16 WMMA GEMM with LDS-staged A tile (async copy, double-buffered):
//   C[M,N] = scale * (A[M,K] * Bt[N,K]^T) (+ epilogue)
// Block = 256 threads = 8 waves; block tile 32(M) x 512(N); wave tile 32x64.
// A tile (32 x Ktile=128) is staged in LDS once per block (8x traffic cut)
// via GLOBAL_LOAD_ASYNC_TO_LDS_B128 when available; K-chunk double buffering.
// grid = (tilesN/8, tilesM, batch).
//
// EPI 0: fp32 out, scale only
// EPI 1: fp32 out, + bias[n]
// EPI 3: bf16 out, + add[(m%32)*N + n]   (positional-projection epilogue)
// EPI 4: bf16 out, no add
// ---------------------------------------------------------------------------
#define KTILE 128
#define APAD  8                    // row stride 136 elems = 272B -> bank 4r, conflict-free
#define AROW  (KTILE + APAD)

template <int EPI>
__global__ __launch_bounds__(256) void
k_gemm_nt(const __bf16* __restrict__ A, const __bf16* __restrict__ Bt,
          void* __restrict__ Cout, const float* __restrict__ add,
          int Mrows, int Ncols, int Kdim, float scale,
          long long sA, long long sB, long long sC) {
  __shared__ __bf16 As[2][32 * AROW];

  const int bz = blockIdx.z;
  A  += (size_t)bz * (size_t)sA;
  Bt += (size_t)bz * (size_t)sB;

  const int m0 = blockIdx.y << 5;                      // block-uniform M tile
  const int tid = threadIdx.x;
  const int waveid = tid >> 5;
  const int n0 = (blockIdx.x * 8 + waveid) << 6;       // wave's N tile

  // Cooperative A-chunk stage: 32 rows x 128 elems x 2B = 8 KB = 512 x 16B segs.
  auto issue_chunk = [&](int k0, int buf) {
#pragma unroll
    for (int it = 0; it < 2; ++it) {
      int s = tid + it * 256;
      int row = s >> 4, c16 = s & 15;
      const __bf16* g = A + (size_t)(m0 + row) * Kdim + k0 + c16 * 8;
      __bf16* l = &As[buf][row * AROW + c16 * 8];
#if HAS_ASYNC
      __builtin_amdgcn_global_load_async_to_lds_b128(
          (AS1 v4i*)(__UINTPTR_TYPE__)g,
          (AS3 v4i*)(unsigned)(__UINTPTR_TYPE__)l, 0, 0);
#else
      *(v8bf*)l = *(const v8bf*)g;
#endif
    }
  };

  const int lane = tid & 31;
  const int half = lane >> 4;       // 0: lanes 0-15, 1: lanes 16-31
  const int r    = lane & 15;

  // B fragment (32x16 bf16, NT): lane holds column n0+j*16+r,
  // elems 0..15 = K[16*half .. +16)  (contiguous 32B load)
  const __bf16* b0p = Bt + (size_t)(n0 + r) * Kdim + half * 16;
  const __bf16* b1p = b0p + (size_t)16 * Kdim;
  const __bf16* b2p = b0p + (size_t)32 * Kdim;
  const __bf16* b3p = b0p + (size_t)48 * Kdim;

  v8f acc[2][4] = {};

  issue_chunk(0, 0);
  const int nchunks = Kdim / KTILE;

  for (int c = 0; c < nchunks; ++c) {
    const int buf = c & 1;
#if HAS_ASYNC
    __builtin_amdgcn_s_wait_asynccnt(0);
#endif
    __syncthreads();                               // chunk c resident; all reads of buf done
    if (c + 1 < nchunks) issue_chunk((c + 1) * KTILE, buf ^ 1);

    const __bf16* Ab = &As[buf][0];
    const int kg = c * KTILE;
#pragma unroll
    for (int kk = 0; kk < KTILE; kk += 32) {
      // A frags from LDS: lane row r (frag0) / r+16 (frag1);
      // elems 0..15 = K[kk+half*16 .. +16)
      const __bf16* ar0 = Ab + r * AROW + kk + half * 16;
      const __bf16* ar1 = Ab + (r + 16) * AROW + kk + half * 16;
      v8bf al0 = *(const v8bf*)ar0;
      v8bf ah0 = *(const v8bf*)(ar0 + 8);
      v8bf al1 = *(const v8bf*)ar1;
      v8bf ah1 = *(const v8bf*)(ar1 + 8);
      v16bf a0 = __builtin_shufflevector(al0, ah0, 0,1,2,3,4,5,6,7,8,9,10,11,12,13,14,15);
      v16bf a1 = __builtin_shufflevector(al1, ah1, 0,1,2,3,4,5,6,7,8,9,10,11,12,13,14,15);

      const int k = kg + kk;
      v16bf b0 = *(const v16bf*)(b0p + k);
      v16bf b1 = *(const v16bf*)(b1p + k);
      v16bf b2 = *(const v16bf*)(b2p + k);
      v16bf b3 = *(const v16bf*)(b3p + k);

      acc[0][0] = __builtin_amdgcn_wmma_f32_16x16x32_bf16(false, a0, false, b0, (short)0, acc[0][0], false, false);
      acc[0][1] = __builtin_amdgcn_wmma_f32_16x16x32_bf16(false, a0, false, b1, (short)0, acc[0][1], false, false);
      acc[0][2] = __builtin_amdgcn_wmma_f32_16x16x32_bf16(false, a0, false, b2, (short)0, acc[0][2], false, false);
      acc[0][3] = __builtin_amdgcn_wmma_f32_16x16x32_bf16(false, a0, false, b3, (short)0, acc[0][3], false, false);
      acc[1][0] = __builtin_amdgcn_wmma_f32_16x16x32_bf16(false, a1, false, b0, (short)0, acc[1][0], false, false);
      acc[1][1] = __builtin_amdgcn_wmma_f32_16x16x32_bf16(false, a1, false, b1, (short)0, acc[1][1], false, false);
      acc[1][2] = __builtin_amdgcn_wmma_f32_16x16x32_bf16(false, a1, false, b2, (short)0, acc[1][2], false, false);
      acc[1][3] = __builtin_amdgcn_wmma_f32_16x16x32_bf16(false, a1, false, b3, (short)0, acc[1][3], false, false);
    }
  }

  // Epilogue. C/D layout: VGPR rr: lanes0-15 -> row rr, lanes16-31 -> row rr+8; col = lane&15.
  float* Cf = nullptr;
  __bf16* Cb = nullptr;
  if constexpr (EPI == 0 || EPI == 1) Cf = (float*)Cout + (size_t)bz * (size_t)sC;
  else                                Cb = (__bf16*)Cout + (size_t)bz * (size_t)sC;

  const int rbase = half * 8;
#pragma unroll
  for (int i = 0; i < 2; ++i) {
#pragma unroll
    for (int j = 0; j < 4; ++j) {
      const int n = n0 + j * 16 + r;
#pragma unroll
      for (int rr = 0; rr < 8; ++rr) {
        const int m = m0 + i * 16 + rbase + rr;
        float v = acc[i][j][rr] * scale;
        if constexpr (EPI == 1) v += add[n];
        if constexpr (EPI == 3) v += add[(size_t)(m & (CC - 1)) * Ncols + n];
        if constexpr (EPI == 0 || EPI == 1) Cf[(size_t)m * Ncols + n] = v;
        else                                Cb[(size_t)m * Ncols + n] = f2bf(v);
      }
    }
  }
}

// ---------------------------------------------------------------------------
// Top-K (K=8) over M=1024 logits per (b,q) row + softmax weights.
// One wave32 per row; 8 iterative wave-argmax passes with shfl_xor reduction.
// ---------------------------------------------------------------------------
__global__ void k_topk(const float* __restrict__ logits, const unsigned char* __restrict__ mask,
                       float* __restrict__ topw, int* __restrict__ topi) {
  const int bq = blockIdx.x;
  const int lane = threadIdx.x;
  const float* row = logits + (size_t)bq * MM;
  const unsigned char* mrow = mask + (size_t)bq * MM;

  float vals[MM / 32];
#pragma unroll
  for (int j = 0; j < MM / 32; ++j) {
    int i = j * 32 + lane;
    float v = row[i];
    if (!mrow[i]) v = -1.0e6f;
    vals[j] = v;
  }

  float sv[KTOP];
  int   si[KTOP];
#pragma unroll
  for (int t = 0; t < KTOP; ++t) {
    float bv = -3.4e38f;
    int bi = 0x3fffffff;
#pragma unroll
    for (int j = 0; j < MM / 32; ++j) {
      if (vals[j] > bv) { bv = vals[j]; bi = j * 32 + lane; }
    }
    for (int off = 16; off; off >>= 1) {
      float ov = __shfl_xor(bv, off);
      int   oi = __shfl_xor(bi, off);
      if (ov > bv || (ov == bv && oi < bi)) { bv = ov; bi = oi; }
    }
    sv[t] = bv; si[t] = bi;
    if ((bi & 31) == lane) vals[bi >> 5] = -3.4e38f;  // exclude winner
  }

  float mx = sv[0];
#pragma unroll
  for (int t = 1; t < KTOP; ++t) mx = fmaxf(mx, sv[t]);
  float e[KTOP], s = 0.f;
#pragma unroll
  for (int t = 0; t < KTOP; ++t) { e[t] = __expf(sv[t] - mx); s += e[t]; }
  if (lane < KTOP) {
    topw[(size_t)bq * KTOP + lane] = e[lane] / s;
    topi[(size_t)bq * KTOP + lane] = si[lane];
  }
}

// ---------------------------------------------------------------------------
// Per-(b,q) multi-head attention over top-K gathered chunks.
// Block 256 = 8 waves; wave w handles head h=w; lane = chunk row c (C=32).
// kproj/vproj are precomputed bf16 [B*M, C, F] (pos-projection folded in).
// Output: ctx[b*Q+q, F] (bf16) = sum_k weight_k * o_k   (pre-Wo).
// ---------------------------------------------------------------------------
__global__ void k_attention(const float* __restrict__ qh,       // [B*Q, F] fp32 (q proj + bias)
                            const __bf16* __restrict__ kproj,   // [B*M*C, F]
                            const __bf16* __restrict__ vproj,   // [B*M*C, F]
                            const float* __restrict__ topw,     // [B*Q, 8]
                            const int* __restrict__ topi,       // [B*Q, 8]
                            __bf16* __restrict__ ctx) {         // [B*Q, F]
  __shared__ float qs[FF];
  const int bq = blockIdx.x;
  const int tid = threadIdx.x;
  const int b = bq >> 8;                 // Q = 256
  qs[tid]       = qh[(size_t)bq * FF + tid];
  qs[tid + 256] = qh[(size_t)bq * FF + tid + 256];
  __syncthreads();

  const int h = tid >> 5;                // head
  const int lane = tid & 31;             // chunk row c
  const float* qsh = qs + h * HDIM;

  float accA = 0.f, accB = 0.f;

  for (int k = 0; k < KTOP; ++k) {
    const int m = topi[(size_t)bq * KTOP + k];
    const float wgt = topw[(size_t)bq * KTOP + k];
    const size_t chunk = ((size_t)b * MM + (size_t)m) * (size_t)(CC * FF);

    // logit for (h, c=lane): dot over HD=64, scaled by 1/sqrt(64)
    const v8bf* kr = (const v8bf*)(kproj + chunk + (size_t)lane * FF + h * HDIM);
    float lg = 0.f;
#pragma unroll
    for (int t = 0; t < 8; ++t) {
      v8bf kv = kr[t];
#pragma unroll
      for (int e2 = 0; e2 < 8; ++e2) lg += qsh[t * 8 + e2] * bf2f(kv[e2]);
    }
    lg *= 0.125f;

    // softmax over C=32 (one wave)
    float mx = lg;
    for (int off = 16; off; off >>= 1) mx = fmaxf(mx, __shfl_xor(mx, off));
    float ex = __expf(lg - mx);
    float sm = ex;
    for (int off = 16; off; off >>= 1) sm += __shfl_xor(sm, off);
    const float attn = ex / sm;

    // o[h, hd] = sum_c attn[c] * v[c, h*64+hd]; lane owns hd pair (2*lane, 2*lane+1)
    const __bf16* vb = vproj + chunk + (size_t)h * HDIM + (size_t)lane * 2;
    float o0 = 0.f, o1 = 0.f;
#pragma unroll
    for (int c = 0; c < CC; ++c) {
      const float a = __shfl(attn, c);
      const __bf16* vr = vb + (size_t)c * FF;
      o0 += a * bf2f(vr[0]);
      o1 += a * bf2f(vr[1]);
    }
    accA += wgt * o0;
    accB += wgt * o1;
  }

  __bf16* crow = ctx + (size_t)bq * FF + h * HDIM + lane * 2;
  crow[0] = f2bf(accA);
  crow[1] = f2bf(accB);
}

// ---------------------------------------------------------------------------
// Launch
// ---------------------------------------------------------------------------
static inline dim3 gemm_grid(int Mrows, int Ncols, int batches) {
  return dim3((unsigned)((Ncols >> 6) / 8), (unsigned)(Mrows >> 5), (unsigned)batches);
}

extern "C" void kernel_launch(void* const* d_in, const int* in_sizes, int n_in,
                              void* d_out, int out_size, void* d_ws, size_t ws_size,
                              hipStream_t stream) {
  const float* queries  = (const float*)d_in[0];   // [B,Q,E]
  const float* mem_keys = (const float*)d_in[1];   // [B,M,D]
  const float* mem_cont = (const float*)d_in[2];   // [B,M,C,D]
  const unsigned char* hm_mask = (const unsigned char*)d_in[3]; // [B,Q,M] bool
  const float* Wq     = (const float*)d_in[4];     // [E,F]
  const float* Wk     = (const float*)d_in[5];     // [D,F]
  const float* mha_wq = (const float*)d_in[6];     // [E,F]
  const float* mha_bq = (const float*)d_in[7];     // [F]
  const float* mha_wk = (const float*)d_in[8];     // [D,F]
  const float* mha_bk = (const float*)d_in[9];     // [F]
  const float* mha_wv = (const float*)d_in[10];    // [D,F]
  const float* mha_bv = (const float*)d_in[11];    // [F]
  const float* mha_wo = (const float*)d_in[12];    // [F,F]
  const float* mha_bo = (const float*)d_in[13];    // [F]
  float* out = (float*)d_out;                      // [B,Q,F]

  // ---- workspace bump allocator (256B aligned) ----
  char* p = (char*)d_ws;
  auto alloc = [&](size_t bytes) -> char* {
    char* ret = p;
    p += (bytes + 255) & ~(size_t)255;
    return ret;
  };
  __bf16* WqT   = (__bf16*)alloc((size_t)FF * EE * 2);
  __bf16* WkT   = (__bf16*)alloc((size_t)FF * DD * 2);
  __bf16* mwqT  = (__bf16*)alloc((size_t)FF * EE * 2);
  __bf16* mwkT  = (__bf16*)alloc((size_t)FF * DD * 2);
  __bf16* mwvT  = (__bf16*)alloc((size_t)FF * DD * 2);
  __bf16* mwoT  = (__bf16*)alloc((size_t)FF * FF * 2);
  __bf16* q_bf  = (__bf16*)alloc((size_t)BB * QQ * EE * 2);
  __bf16* keys_bf = (__bf16*)alloc((size_t)BB * MM * DD * 2);
  __bf16* cont_bf = (__bf16*)alloc((size_t)BB * MM * CC * DD * 2);
  __bf16* pos_bf  = (__bf16*)alloc((size_t)CC * DD * 2);
  float*  posK    = (float*)alloc((size_t)CC * FF * 4);
  float*  posV    = (float*)alloc((size_t)CC * FF * 4);
  __bf16* qh_bf   = (__bf16*)alloc((size_t)BB * QQ * FF * 2);
  __bf16* kh_bf   = (__bf16*)alloc((size_t)BB * MM * FF * 2);
  float*  q_h     = (float*)alloc((size_t)BB * QQ * FF * 4);
  float*  logits  = (float*)alloc((size_t)BB * QQ * MM * 4);
  float*  topw    = (float*)alloc((size_t)BB * QQ * KTOP * 4);
  int*    topi    = (int*)alloc((size_t)BB * QQ * KTOP * 4);
  __bf16* kproj   = (__bf16*)alloc((size_t)BB * MM * CC * FF * 2);
  __bf16* vproj   = (__bf16*)alloc((size_t)BB * MM * CC * FF * 2);
  __bf16* ctx_bf  = (__bf16*)alloc((size_t)BB * QQ * FF * 2);

  const float inv_sqrt_F = 0.044194173824159216f;  // 1/sqrt(512)

  // 1) weight transpose-convert (fp32 [K,N] -> bf16 [N,K])
  {
    dim3 g(512 * 512 / 256), t(256);
    k_transpose_bf16<<<g, t, 0, stream>>>(Wq,     WqT,  EE, FF);
    k_transpose_bf16<<<g, t, 0, stream>>>(Wk,     WkT,  DD, FF);
    k_transpose_bf16<<<g, t, 0, stream>>>(mha_wq, mwqT, EE, FF);
    k_transpose_bf16<<<g, t, 0, stream>>>(mha_wk, mwkT, DD, FF);
    k_transpose_bf16<<<g, t, 0, stream>>>(mha_wv, mwvT, DD, FF);
    k_transpose_bf16<<<g, t, 0, stream>>>(mha_wo, mwoT, FF, FF);
  }

  // 2) activation conversions to bf16
  {
    size_t nq = (size_t)BB * QQ * EE;
    size_t nk = (size_t)BB * MM * DD;
    size_t nc = (size_t)BB * MM * CC * DD;
    k_f32_to_bf16<<<dim3((unsigned)((nq + 255) / 256)), 256, 0, stream>>>(queries, q_bf, nq);
    k_f32_to_bf16<<<dim3((unsigned)((nk + 255) / 256)), 256, 0, stream>>>(mem_keys, keys_bf, nk);
    k_f32_to_bf16<<<dim3(131072), 256, 0, stream>>>(mem_cont, cont_bf, nc);
  }

  // 3) positional encodings + their K/V projections (folds pos+bias into epilogue adds)
  k_pos_encode<<<dim3(CC * DD / 256), 256, 0, stream>>>(pos_bf);
  k_gemm_nt<1><<<gemm_grid(CC, FF, 1), 256, 0, stream>>>(pos_bf, mwkT, posK, mha_bk, CC, FF, DD, 1.0f, 0, 0, 0);
  k_gemm_nt<1><<<gemm_grid(CC, FF, 1), 256, 0, stream>>>(pos_bf, mwvT, posV, mha_bv, CC, FF, DD, 1.0f, 0, 0, 0);

  // 4) retrieval projections
  k_gemm_nt<4><<<gemm_grid(BB * QQ, FF, 1), 256, 0, stream>>>(q_bf, WqT, qh_bf, nullptr, BB * QQ, FF, EE, 1.0f, 0, 0, 0);
  k_gemm_nt<4><<<gemm_grid(BB * MM, FF, 1), 256, 0, stream>>>(keys_bf, WkT, kh_bf, nullptr, BB * MM, FF, DD, 1.0f, 0, 0, 0);

  // 5) MHA query projection (+bias), fp32 out
  k_gemm_nt<1><<<gemm_grid(BB * QQ, FF, 1), 256, 0, stream>>>(q_bf, mwqT, q_h, mha_bq, BB * QQ, FF, EE, 1.0f, 0, 0, 0);

  // 6) retrieval logits: batched NT GEMM  [Q,F] x [M,F]^T / sqrt(F)
  k_gemm_nt<0><<<gemm_grid(QQ, MM, BB), 256, 0, stream>>>(
      qh_bf, kh_bf, logits, nullptr, QQ, MM, FF, inv_sqrt_F,
      (long long)QQ * FF, (long long)MM * FF, (long long)QQ * MM);

  // 7) top-k + softmax weights (one wave per (b,q))
  k_topk<<<dim3(BB * QQ), 32, 0, stream>>>(logits, hm_mask, topw, topi);

  // 8) precompute K/V projections for ALL chunks: [B*M*C, D] x [D,F] + pos-proj row add
  k_gemm_nt<3><<<gemm_grid(BB * MM * CC, FF, 1), 256, 0, stream>>>(
      cont_bf, mwkT, kproj, posK, BB * MM * CC, FF, DD, 1.0f, 0, 0, 0);
  k_gemm_nt<3><<<gemm_grid(BB * MM * CC, FF, 1), 256, 0, stream>>>(
      cont_bf, mwvT, vproj, posV, BB * MM * CC, FF, DD, 1.0f, 0, 0, 0);

  // 9) per-(b,q) attention over gathered top-k chunks, weighted combine over k
  k_attention<<<dim3(BB * QQ), 256, 0, stream>>>(q_h, kproj, vproj, topw, topi, ctx_bf);

  // 10) output projection: out = ctx @ Wo + bo   (weights sum to 1 => bias once)
  k_gemm_nt<1><<<gemm_grid(BB * QQ, FF, 1), 256, 0, stream>>>(
      ctx_bf, mwoT, out, mha_bo, BB * QQ, FF, FF, 1.0f, 0, 0, 0);
}
